// GPTLanguageModelWithEncoder_41884521071045
// MI455X (gfx1250) — matrix operand order repaired and checked by reading, compiled, not run on hardware
//
#include <hip/hip_runtime.h>
#include <math.h>

typedef __attribute__((ext_vector_type(16))) _Float16 v16h;
typedef __attribute__((ext_vector_type(8)))  float    v8f;

union Frag {              // 32 bytes: one WMMA 16-bit operand fragment
    v16h     v;
    _Float16 h[16];
    float4   f4[2];
};

#define L_  6
#define D_  768
#define H_  12
#define HD_ 64
#define V_  32000
#define B_  4
#define T_  1024
#define FF_ 3072
#define BT_ (B_ * T_)

// pack 4 fp32 -> 4 f16 and store as one 8-byte LDS write
__device__ __forceinline__ void cvt4(_Float16* dst, float4 v) {
    union { _Float16 h[4]; uint2 u; } t;
    t.h[0] = (_Float16)v.x; t.h[1] = (_Float16)v.y;
    t.h[2] = (_Float16)v.z; t.h[3] = (_Float16)v.w;
    *reinterpret_cast<uint2*>(dst) = t.u;
}

// ---------------- embedding: out[b,t,:] = tok[idx[b,t],:] + pos[t,:] ----------------
__global__ void embed_kernel(const int* __restrict__ idx, const float* __restrict__ tok,
                             const float* __restrict__ pos, float* __restrict__ out) {
    int row = blockIdx.x;                 // 0..BT-1
    int t = row % T_;
    int token = idx[row];
    const float4* tr = reinterpret_cast<const float4*>(tok + (long)token * D_);
    const float4* pr = reinterpret_cast<const float4*>(pos + (long)t * D_);
    float4* orow = reinterpret_cast<float4*>(out + (long)row * D_);
    for (int d = threadIdx.x; d < D_ / 4; d += blockDim.x) {
        float4 a = tr[d], b = pr[d];
        float4 o; o.x = a.x + b.x; o.y = a.y + b.y; o.z = a.z + b.z; o.w = a.w + b.w;
        orow[d] = o;
    }
}

// ---------------- layernorm over D=768, one 256-thread block per row ----------------
__global__ __launch_bounds__(256)
void ln_kernel(const float* __restrict__ x, const float* __restrict__ g,
               const float* __restrict__ b, float* __restrict__ out) {
    __shared__ float red[256];
    int row = blockIdx.x;
    int tid = threadIdx.x;
    const float* xr = x + (long)row * D_;
    float v0 = xr[tid], v1 = xr[tid + 256], v2 = xr[tid + 512];
    red[tid] = v0 + v1 + v2;
    __syncthreads();
    for (int o = 128; o > 0; o >>= 1) {
        if (tid < o) red[tid] += red[tid + o];
        __syncthreads();
    }
    float mean = red[0] * (1.0f / D_);
    __syncthreads();
    float d0 = v0 - mean, d1 = v1 - mean, d2 = v2 - mean;
    red[tid] = d0 * d0 + d1 * d1 + d2 * d2;
    __syncthreads();
    for (int o = 128; o > 0; o >>= 1) {
        if (tid < o) red[tid] += red[tid + o];
        __syncthreads();
    }
    float rstd = rsqrtf(red[0] * (1.0f / D_) + 1e-5f);
    float* orow = out + (long)row * D_;
    orow[tid]       = d0 * rstd * g[tid]       + b[tid];
    orow[tid + 256] = d1 * rstd * g[tid + 256] + b[tid + 256];
    orow[tid + 512] = d2 * rstd * g[tid + 512] + b[tid + 512];
}

// ---------------- GEMM: C[4096,N] = A[4096,K] @ W[K,N] (+bias)(+residual)(ReLU?) ----------------
// N, K compile-time. 128x64 block tile, 8 waves, 4 WMMAs per wave per k-step.
// Register prefetch of the next tile + LDS ping-pong -> one barrier per k-step,
// global loads overlap the WMMAs. STREAM=true uses nontemporal output stores.
template<int N, int K, bool STREAM>
__global__ __launch_bounds__(256)
void gemm_wmma(const float* __restrict__ A, const float* __restrict__ W,
               const float* __restrict__ bias, const float* __restrict__ res,
               float* __restrict__ C, int relu) {
    __shared__ _Float16 As[2][128][40];   // [m][k], 80B stride (16B aligned)
    __shared__ _Float16 BsT[2][64][40];   // transposed [n][k]
    int tid  = threadIdx.x;
    int lane = tid & 31, wave = tid >> 5;
    int hx = lane >> 4, r = lane & 15;
    int m0 = blockIdx.y * 128, n0 = blockIdx.x * 64;
    v8f acc[4] = {{}, {}, {}, {}};
    int sa_col = (tid & 7) * 4;      // 0..28
    int sa_row = tid >> 3;           // 0..31 (+32*u)
    int sb_c   = tid & 63;           // output column
    int sb_r0  = (tid >> 6) * 8;     // k group

    const float* Ap = A + (long)(m0 + sa_row) * K + sa_col;  // + 32u*K + kk
    const float* Wp = W + (long)sb_r0 * N + n0 + sb_c;       // + u*N + kk*N

    float4 pa[4];
    float  pb[8];
    // prologue: fetch tile kk=0, commit to buffer 0
#pragma unroll
    for (int u = 0; u < 4; ++u) pa[u] = *reinterpret_cast<const float4*>(Ap + (long)32 * u * K);
#pragma unroll
    for (int u = 0; u < 8; ++u) pb[u] = Wp[(long)u * N];
#pragma unroll
    for (int u = 0; u < 4; ++u) cvt4(&As[0][sa_row + 32 * u][sa_col], pa[u]);
    {
        union { _Float16 h[8]; float4 f; } t;
#pragma unroll
        for (int u = 0; u < 8; ++u) t.h[u] = (_Float16)pb[u];
        *reinterpret_cast<float4*>(&BsT[0][sb_c][sb_r0]) = t.f;
    }
    __syncthreads();

    int p = 0;
#pragma unroll 1
    for (int kk = 0; kk < K; kk += 32) {
        bool more = (kk + 32 < K);
        // issue next tile's global loads; they overlap the WMMAs below
        if (more) {
#pragma unroll
            for (int u = 0; u < 4; ++u)
                pa[u] = *reinterpret_cast<const float4*>(Ap + (long)32 * u * K + (kk + 32));
#pragma unroll
            for (int u = 0; u < 8; ++u)
                pb[u] = Wp[(long)u * N + (long)(kk + 32) * N];
        }
        Frag a;
        a.f4[0] = *reinterpret_cast<const float4*>(&As[p][wave * 16 + r][16 * hx]);
        a.f4[1] = *reinterpret_cast<const float4*>(&As[p][wave * 16 + r][16 * hx + 8]);
#pragma unroll
        for (int t = 0; t < 4; ++t) {
            Frag b;
            b.f4[0] = *reinterpret_cast<const float4*>(&BsT[p][t * 16 + r][16 * hx]);
            b.f4[1] = *reinterpret_cast<const float4*>(&BsT[p][t * 16 + r][16 * hx + 8]);
            acc[t] = __builtin_amdgcn_wmma_f32_16x16x32_f16(false, a.v, false, b.v,
                                                            (short)0, acc[t], false, false);
        }
        // commit prefetched tile to the other buffer (no reader conflict)
        if (more) {
#pragma unroll
            for (int u = 0; u < 4; ++u) cvt4(&As[p ^ 1][sa_row + 32 * u][sa_col], pa[u]);
            union { _Float16 h[8]; float4 f; } t;
#pragma unroll
            for (int u = 0; u < 8; ++u) t.h[u] = (_Float16)pb[u];
            *reinterpret_cast<float4*>(&BsT[p ^ 1][sb_c][sb_r0]) = t.f;
        }
        __syncthreads();
        p ^= 1;
    }
    // epilogue: C layout -> VGPR j is row (j + 8*hx), lane r is column
#pragma unroll
    for (int t = 0; t < 4; ++t) {
        int gc = n0 + t * 16 + r;
        float bs = bias ? bias[gc] : 0.f;
#pragma unroll
        for (int j = 0; j < 8; ++j) {
            int gr = m0 + wave * 16 + 8 * hx + j;
            float o = acc[t][j] + bs;
            if (res)  o += res[(long)gr * N + gc];
            if (relu) o = fmaxf(o, 0.f);
            if (STREAM) __builtin_nontemporal_store(o, &C[(long)gr * N + gc]);
            else        C[(long)gr * N + gc] = o;
        }
    }
}

// ---------------- fused flash-style attention ----------------
// grid (T/64, H, B), block 128 (4 waves); wave handles 16 q-rows, HD=64.
// Kl row-major [key][hd] -> score B frag contiguous; VlT transposed [hd][key]
// -> PV B frag contiguous. K/V register prefetch + LDS ping-pong (one barrier
// per key tile); P round-trip uses wave-local s_wait_dscnt. Mask compares only
// run on diagonal tiles (wave-uniform branch).
template<int CAUSAL>
__global__ __launch_bounds__(128)
void attn_wmma(const float* __restrict__ Q, const float* __restrict__ Kb,
               const float* __restrict__ Vb, float* __restrict__ O) {
    __shared__ _Float16 Kl[2][32][72];
    __shared__ _Float16 VlT[2][64][40];
    __shared__ _Float16 Pl[4][16][40];
    int bz = blockIdx.z, h = blockIdx.y;
    int tid = threadIdx.x;
    int lane = tid & 31, wave = tid >> 5;
    int hx = lane >> 4, r = lane & 15;
    long base = (long)bz * T_ * D_ + h * HD_;
    const float* Qp = Q + base;
    const float* Kp = Kb + base;
    const float* Vp = Vb + base;
    float* Op = O + base;
    int qbase = blockIdx.x * 64 + wave * 16;

    // Q fragments via float4 global loads, pre-scaled by HD^-0.5
    Frag qa0, qa1;
    {
        const float* qrow = Qp + (long)(qbase + r) * D_;
        const float s = 0.125f;
#pragma unroll
        for (int g = 0; g < 2; ++g) {          // g=0 -> qa0 (K 0..31), g=1 -> qa1 (K 32..63)
            Frag& qf = g ? qa1 : qa0;
            const float* p = qrow + 32 * g;
            float4 a0 = *reinterpret_cast<const float4*>(p + 16 * hx);
            float4 a1 = *reinterpret_cast<const float4*>(p + 16 * hx + 4);
            float4 b0 = *reinterpret_cast<const float4*>(p + 16 * hx + 8);
            float4 b1 = *reinterpret_cast<const float4*>(p + 16 * hx + 12);
            qf.h[0] = (_Float16)(a0.x * s); qf.h[1] = (_Float16)(a0.y * s);
            qf.h[2] = (_Float16)(a0.z * s); qf.h[3] = (_Float16)(a0.w * s);
            qf.h[4] = (_Float16)(a1.x * s); qf.h[5] = (_Float16)(a1.y * s);
            qf.h[6] = (_Float16)(a1.z * s); qf.h[7] = (_Float16)(a1.w * s);
            qf.h[8]  = (_Float16)(b0.x * s); qf.h[9]  = (_Float16)(b0.y * s);
            qf.h[10] = (_Float16)(b0.z * s); qf.h[11] = (_Float16)(b0.w * s);
            qf.h[12] = (_Float16)(b1.x * s); qf.h[13] = (_Float16)(b1.y * s);
            qf.h[14] = (_Float16)(b1.z * s); qf.h[15] = (_Float16)(b1.w * s);
        }
    }
    v8f acc0 = {}, acc1 = {}, acc2 = {}, acc3 = {};
    float mrow[8], lrow[8];
#pragma unroll
    for (int j = 0; j < 8; ++j) { mrow[j] = -1e30f; lrow[j] = 0.f; }

    // staging indices
    int kc4  = (tid & 15) * 4;   // hd column group for K
    int krow = tid >> 4;         // key 0..7 (+8u)
    int vhd  = tid & 63;         // hd column for V
    int vk0  = (tid >> 6) * 16;  // key group for V

    float4 pk[4];
    float  pv[16];
    int nkt = CAUSAL ? (blockIdx.x * 2 + 2) : (T_ / 32);
    // prologue: fetch tile 0, commit to buffer 0
#pragma unroll
    for (int u = 0; u < 4; ++u)
        pk[u] = *reinterpret_cast<const float4*>(&Kp[(long)(krow + 8 * u) * D_ + kc4]);
#pragma unroll
    for (int u = 0; u < 16; ++u)
        pv[u] = Vp[(long)(vk0 + u) * D_ + vhd];
#pragma unroll
    for (int u = 0; u < 4; ++u) cvt4(&Kl[0][krow + 8 * u][kc4], pk[u]);
    {
        union { _Float16 h[16]; float4 f4[2]; } t;
#pragma unroll
        for (int u = 0; u < 16; ++u) t.h[u] = (_Float16)pv[u];
        *reinterpret_cast<float4*>(&VlT[0][vhd][vk0])     = t.f4[0];
        *reinterpret_cast<float4*>(&VlT[0][vhd][vk0 + 8]) = t.f4[1];
    }
    __syncthreads();

    int p = 0;
#pragma unroll 1
    for (int kt = 0; kt < nkt; ++kt) {
        bool more = (kt + 1 < nkt);
        // issue next tile's global loads (overlap with score/softmax/PV work)
        if (more) {
            long koff = (long)(kt + 1) * 32 * D_;
#pragma unroll
            for (int u = 0; u < 4; ++u)
                pk[u] = *reinterpret_cast<const float4*>(&Kp[koff + (long)(krow + 8 * u) * D_ + kc4]);
#pragma unroll
            for (int u = 0; u < 16; ++u)
                pv[u] = Vp[koff + (long)(vk0 + u) * D_ + vhd];
        }

        // S = Q @ K^T for two 16-key sub-tiles (B frag: K-dim = hd, N = key)
        v8f s0 = {}, s1 = {};
        Frag bk;
        bk.f4[0] = *reinterpret_cast<const float4*>(&Kl[p][r][16 * hx]);
        bk.f4[1] = *reinterpret_cast<const float4*>(&Kl[p][r][16 * hx + 8]);
        s0 = __builtin_amdgcn_wmma_f32_16x16x32_f16(false, qa0.v, false, bk.v, (short)0, s0, false, false);
        bk.f4[0] = *reinterpret_cast<const float4*>(&Kl[p][r][32 + 16 * hx]);
        bk.f4[1] = *reinterpret_cast<const float4*>(&Kl[p][r][32 + 16 * hx + 8]);
        s0 = __builtin_amdgcn_wmma_f32_16x16x32_f16(false, qa1.v, false, bk.v, (short)0, s0, false, false);
        bk.f4[0] = *reinterpret_cast<const float4*>(&Kl[p][16 + r][16 * hx]);
        bk.f4[1] = *reinterpret_cast<const float4*>(&Kl[p][16 + r][16 * hx + 8]);
        s1 = __builtin_amdgcn_wmma_f32_16x16x32_f16(false, qa0.v, false, bk.v, (short)0, s1, false, false);
        bk.f4[0] = *reinterpret_cast<const float4*>(&Kl[p][16 + r][32 + 16 * hx]);
        bk.f4[1] = *reinterpret_cast<const float4*>(&Kl[p][16 + r][32 + 16 * hx + 8]);
        s1 = __builtin_amdgcn_wmma_f32_16x16x32_f16(false, qa1.v, false, bk.v, (short)0, s1, false, false);

        // online softmax; mask compares only on diagonal tiles (uniform branch)
        auto softmax_step = [&](bool masked) {
#pragma unroll
            for (int j = 0; j < 8; ++j) {
                int qg = qbase + 8 * hx + j;
                float v0 = s0[j], v1 = s1[j];
                if (masked) {
                    if (kt * 32 + r > qg)      v0 = -1e30f;
                    if (kt * 32 + 16 + r > qg) v1 = -1e30f;
                }
                float tmax = fmaxf(v0, v1);
#pragma unroll
                for (int o = 8; o >= 1; o >>= 1) tmax = fmaxf(tmax, __shfl_xor(tmax, o, 16));
                float mnew  = fmaxf(mrow[j], tmax);
                float p0, p1;
                if (masked) {
                    p0 = (v0 < -1e29f) ? 0.f : __expf(v0 - mnew);
                    p1 = (v1 < -1e29f) ? 0.f : __expf(v1 - mnew);
                } else {
                    p0 = __expf(v0 - mnew);
                    p1 = __expf(v1 - mnew);
                }
                float alpha = __expf(mrow[j] - mnew);
                float psum  = p0 + p1;
#pragma unroll
                for (int o = 8; o >= 1; o >>= 1) psum += __shfl_xor(psum, o, 16);
                lrow[j] = lrow[j] * alpha + psum;
                mrow[j] = mnew;
                acc0[j] *= alpha; acc1[j] *= alpha; acc2[j] *= alpha; acc3[j] *= alpha;
                Pl[wave][8 * hx + j][r]      = (_Float16)p0;   // C-layout -> A-layout via LDS
                Pl[wave][8 * hx + j][16 + r] = (_Float16)p1;
            }
        };
        if (CAUSAL && (kt * 32 + 31 > qbase)) softmax_step(true);
        else                                  softmax_step(false);

        // wave-local RAW on own P slice: split-counter wait, no block barrier needed
        asm volatile("s_wait_dscnt 0x0" ::: "memory");

        // O += P @ V   (A frag = P 16x32 contiguous; B frags from VlT contiguous)
        Frag pa, bv;
        pa.f4[0] = *reinterpret_cast<const float4*>(&Pl[wave][r][16 * hx]);
        pa.f4[1] = *reinterpret_cast<const float4*>(&Pl[wave][r][16 * hx + 8]);
        bv.f4[0] = *reinterpret_cast<const float4*>(&VlT[p][r][16 * hx]);
        bv.f4[1] = *reinterpret_cast<const float4*>(&VlT[p][r][16 * hx + 8]);
        acc0 = __builtin_amdgcn_wmma_f32_16x16x32_f16(false, pa.v, false, bv.v, (short)0, acc0, false, false);
        bv.f4[0] = *reinterpret_cast<const float4*>(&VlT[p][16 + r][16 * hx]);
        bv.f4[1] = *reinterpret_cast<const float4*>(&VlT[p][16 + r][16 * hx + 8]);
        acc1 = __builtin_amdgcn_wmma_f32_16x16x32_f16(false, pa.v, false, bv.v, (short)0, acc1, false, false);
        bv.f4[0] = *reinterpret_cast<const float4*>(&VlT[p][32 + r][16 * hx]);
        bv.f4[1] = *reinterpret_cast<const float4*>(&VlT[p][32 + r][16 * hx + 8]);
        acc2 = __builtin_amdgcn_wmma_f32_16x16x32_f16(false, pa.v, false, bv.v, (short)0, acc2, false, false);
        bv.f4[0] = *reinterpret_cast<const float4*>(&VlT[p][48 + r][16 * hx]);
        bv.f4[1] = *reinterpret_cast<const float4*>(&VlT[p][48 + r][16 * hx + 8]);
        acc3 = __builtin_amdgcn_wmma_f32_16x16x32_f16(false, pa.v, false, bv.v, (short)0, acc3, false, false);

        // commit next K/V tile to the other buffer (no reader conflict)
        if (more) {
#pragma unroll
            for (int u = 0; u < 4; ++u) cvt4(&Kl[p ^ 1][krow + 8 * u][kc4], pk[u]);
            union { _Float16 h[16]; float4 f4[2]; } t;
#pragma unroll
            for (int u = 0; u < 16; ++u) t.h[u] = (_Float16)pv[u];
            *reinterpret_cast<float4*>(&VlT[p ^ 1][vhd][vk0])     = t.f4[0];
            *reinterpret_cast<float4*>(&VlT[p ^ 1][vhd][vk0 + 8]) = t.f4[1];
        }
        __syncthreads();
        p ^= 1;
    }
    // normalize and write
#pragma unroll
    for (int j = 0; j < 8; ++j) {
        float inv = 1.0f / lrow[j];
        int qg = qbase + 8 * hx + j;
        Op[(long)qg * D_ + r]      = acc0[j] * inv;
        Op[(long)qg * D_ + 16 + r] = acc1[j] * inv;
        Op[(long)qg * D_ + 32 + r] = acc2[j] * inv;
        Op[(long)qg * D_ + 48 + r] = acc3[j] * inv;
    }
}

// templated host-side launcher (M fixed at BT_)
template<int N, int K, bool STREAM = false>
static inline void launch_gemm(const float* A, const float* W, const float* bias,
                               const float* res, float* C, int relu, hipStream_t s) {
    dim3 g(N / 64, BT_ / 128);
    gemm_wmma<N, K, STREAM><<<g, dim3(256), 0, s>>>(A, W, bias, res, C, relu);
}

// ---------------- host orchestration ----------------
extern "C" void kernel_launch(void* const* d_in, const int* in_sizes, int n_in,
                              void* d_out, int out_size, void* d_ws, size_t ws_size,
                              hipStream_t stream) {
    (void)in_sizes; (void)n_in; (void)out_size; (void)ws_size;
    const int*   idx      = (const int*)d_in[0];
    const int*   enc_in   = (const int*)d_in[1];
    const float* enc_tok  = (const float*)d_in[2];
    const float* enc_pos  = (const float*)d_in[3];
    const float* enc_qkv  = (const float*)d_in[4];
    const float* enc_wo   = (const float*)d_in[5];
    const float* enc_bo   = (const float*)d_in[6];
    const float* enc_w1   = (const float*)d_in[7];
    const float* enc_b1   = (const float*)d_in[8];
    const float* enc_w2   = (const float*)d_in[9];
    const float* enc_b2   = (const float*)d_in[10];
    const float* enc_ln1g = (const float*)d_in[11];
    const float* enc_ln1b = (const float*)d_in[12];
    const float* enc_ln2g = (const float*)d_in[13];
    const float* enc_ln2b = (const float*)d_in[14];
    const float* enc_lnfg = (const float*)d_in[15];
    const float* enc_lnfb = (const float*)d_in[16];
    const float* dec_tok  = (const float*)d_in[17];
    const float* dec_pos  = (const float*)d_in[18];
    const float* dec_qkv  = (const float*)d_in[19];
    const float* dec_wo   = (const float*)d_in[20];
    const float* dec_bo   = (const float*)d_in[21];
    const float* dec_cqkv = (const float*)d_in[22];
    const float* dec_cwo  = (const float*)d_in[23];
    const float* dec_cbo  = (const float*)d_in[24];
    const float* dec_w1   = (const float*)d_in[25];
    const float* dec_b1   = (const float*)d_in[26];
    const float* dec_w2   = (const float*)d_in[27];
    const float* dec_b2   = (const float*)d_in[28];
    const float* dec_ln1g = (const float*)d_in[29];
    const float* dec_ln1b = (const float*)d_in[30];
    const float* dec_ln2g = (const float*)d_in[31];
    const float* dec_ln2b = (const float*)d_in[32];
    const float* dec_ln3g = (const float*)d_in[33];
    const float* dec_ln3b = (const float*)d_in[34];
    const float* dec_lnfg = (const float*)d_in[35];
    const float* dec_lnfb = (const float*)d_in[36];
    const float* lm_w     = (const float*)d_in[37];
    const float* lm_b     = (const float*)d_in[38];

    float* ws = (float*)d_ws;
    const size_t NTD = (size_t)BT_ * D_;
    float* xbuf   = ws; ws += NTD;
    float* hbuf   = ws; ws += NTD;
    float* qbuf   = ws; ws += NTD;
    float* kbuf   = ws; ws += NTD;
    float* vbuf   = ws; ws += NTD;
    float* aobuf  = ws; ws += NTD;
    float* encout = ws; ws += NTD;
    float* ffbuf  = ws; ws += (size_t)BT_ * FF_;

    auto ln = [&](const float* x, const float* gg, const float* bb, float* out) {
        ln_kernel<<<dim3(BT_), dim3(256), 0, stream>>>(x, gg, bb, out);
    };
    auto attn = [&](const float* Qp, const float* Kp, const float* Vp, float* Op, int causal) {
        dim3 g(T_ / 64, H_, B_);
        if (causal) attn_wmma<1><<<g, dim3(128), 0, stream>>>(Qp, Kp, Vp, Op);
        else        attn_wmma<0><<<g, dim3(128), 0, stream>>>(Qp, Kp, Vp, Op);
    };

    // ---------------- encoder (causal self-attn, per reference) ----------------
    embed_kernel<<<dim3(BT_), dim3(256), 0, stream>>>(enc_in, enc_tok, enc_pos, xbuf);
    for (int i = 0; i < L_; ++i) {
        ln(xbuf, enc_ln1g + i * D_, enc_ln1b + i * D_, hbuf);
        const float* Wq = enc_qkv + (size_t)(i * 3 + 0) * D_ * D_;
        const float* Wk = enc_qkv + (size_t)(i * 3 + 1) * D_ * D_;
        const float* Wv = enc_qkv + (size_t)(i * 3 + 2) * D_ * D_;
        launch_gemm<D_, D_>(hbuf, Wq, nullptr, nullptr, qbuf, 0, stream);
        launch_gemm<D_, D_>(hbuf, Wk, nullptr, nullptr, kbuf, 0, stream);
        launch_gemm<D_, D_>(hbuf, Wv, nullptr, nullptr, vbuf, 0, stream);
        attn(qbuf, kbuf, vbuf, aobuf, 1);
        launch_gemm<D_, D_>(aobuf, enc_wo + (size_t)i * D_ * D_, enc_bo + i * D_, xbuf, xbuf, 0, stream);
        ln(xbuf, enc_ln2g + i * D_, enc_ln2b + i * D_, hbuf);
        launch_gemm<FF_, D_>(hbuf, enc_w1 + (size_t)i * D_ * FF_, enc_b1 + i * FF_, nullptr, ffbuf, 1, stream);
        launch_gemm<D_, FF_>(ffbuf, enc_w2 + (size_t)i * FF_ * D_, enc_b2 + i * D_, xbuf, xbuf, 0, stream);
    }
    ln(xbuf, enc_lnfg, enc_lnfb, encout);

    // ---------------- decoder ----------------
    embed_kernel<<<dim3(BT_), dim3(256), 0, stream>>>(idx, dec_tok, dec_pos, xbuf);
    for (int i = 0; i < L_; ++i) {
        // causal self-attention
        ln(xbuf, dec_ln1g + i * D_, dec_ln1b + i * D_, hbuf);
        const float* Wq = dec_qkv + (size_t)(i * 3 + 0) * D_ * D_;
        const float* Wk = dec_qkv + (size_t)(i * 3 + 1) * D_ * D_;
        const float* Wv = dec_qkv + (size_t)(i * 3 + 2) * D_ * D_;
        launch_gemm<D_, D_>(hbuf, Wq, nullptr, nullptr, qbuf, 0, stream);
        launch_gemm<D_, D_>(hbuf, Wk, nullptr, nullptr, kbuf, 0, stream);
        launch_gemm<D_, D_>(hbuf, Wv, nullptr, nullptr, vbuf, 0, stream);
        attn(qbuf, kbuf, vbuf, aobuf, 1);
        launch_gemm<D_, D_>(aobuf, dec_wo + (size_t)i * D_ * D_, dec_bo + i * D_, xbuf, xbuf, 0, stream);
        // cross-attention (non-causal, K/V from encoder output)
        ln(xbuf, dec_ln2g + i * D_, dec_ln2b + i * D_, hbuf);
        const float* cWq = dec_cqkv + (size_t)(i * 3 + 0) * D_ * D_;
        const float* cWk = dec_cqkv + (size_t)(i * 3 + 1) * D_ * D_;
        const float* cWv = dec_cqkv + (size_t)(i * 3 + 2) * D_ * D_;
        launch_gemm<D_, D_>(hbuf,   cWq, nullptr, nullptr, qbuf, 0, stream);
        launch_gemm<D_, D_>(encout, cWk, nullptr, nullptr, kbuf, 0, stream);
        launch_gemm<D_, D_>(encout, cWv, nullptr, nullptr, vbuf, 0, stream);
        attn(qbuf, kbuf, vbuf, aobuf, 0);
        launch_gemm<D_, D_>(aobuf, dec_cwo + (size_t)i * D_ * D_, dec_cbo + i * D_, xbuf, xbuf, 0, stream);
        // FFN
        ln(xbuf, dec_ln3g + i * D_, dec_ln3b + i * D_, hbuf);
        launch_gemm<FF_, D_>(hbuf, dec_w1 + (size_t)i * D_ * FF_, dec_b1 + i * FF_, nullptr, ffbuf, 1, stream);
        launch_gemm<D_, FF_>(ffbuf, dec_w2 + (size_t)i * FF_ * D_, dec_b2 + i * D_, xbuf, xbuf, 0, stream);
    }
    ln(xbuf, dec_lnfg, dec_lnfb, hbuf);

    // LM head: logits = y @ lm_w + lm_b -> d_out [B,T,V] fp32 (streaming NT stores)
    launch_gemm<V_, D_, true>(hbuf, lm_w, lm_b, nullptr, (float*)d_out, 0, stream);
}
